// MyLSTM_36309653520672
// MI455X (gfx1250) — matrix operand order repaired and checked by reading, not compiled, not run on hardware
//
#include <hip/hip_runtime.h>
#include <hip/hip_bf16.h>
#include <stdint.h>

// Problem sizes (fixed by reference)
#define Bx   8
#define Sx   128
#define Ex   512
#define Hx   1024
#define Vx   50257
#define Tx   (Bx * Sx)      // 1024 timesteps (LSTM batch of 1)
#define G4H  (4 * Hx)       // 4096 gate width

// ---------- CDNA5 WMMA / TDM types ----------
typedef __bf16        v16bf __attribute__((ext_vector_type(16)));
typedef float         v8f   __attribute__((ext_vector_type(8)));
typedef unsigned int  u32x4 __attribute__((ext_vector_type(4)));
typedef int           i32x4 __attribute__((ext_vector_type(4)));
typedef int           i32x8 __attribute__((ext_vector_type(8)));

union F32x8 { v8f v; float f[8]; };

// ---------- gfx1250 async global->LDS (per-lane addresses; ASYNCcnt) ----------
__device__ __forceinline__ void g2l_16B(void* lds_dst, const float* src) {
  // Low 32 bits of a generic pointer to LDS == LDS byte address.
  unsigned int loff = (unsigned int)(uintptr_t)lds_dst;
  asm volatile("global_load_async_to_lds_b128 %0, %1, off"
               :: "v"(loff), "v"(src) : "memory");
}
__device__ __forceinline__ void wait_async_le2() {
  asm volatile("s_wait_asynccnt 0x2" ::: "memory");
}
__device__ __forceinline__ void wait_async_0() {
  asm volatile("s_wait_asynccnt 0x0" ::: "memory");
}

// ---------- gfx1250 Tensor Data Mover: 2-D tile load, 36-float LDS pitch ----
// D# per CDNA5 ISA: data_size=4B, pad_enable,
// pad_interval=4 (pad after 32 DWORDs), pad_amount=3 (4 DWORDs) -> the same
// 36-float row pitch the WMMA fragment loaders use. OOB rows return zero,
// which natively handles the V edge of W_out.
__device__ __forceinline__ void tdm_load_2d(void* lds_dst, const float* gsrc,
                                            int tile_k, int tile_rows,
                                            unsigned int tensor_k,
                                            unsigned int tensor_rows,
                                            unsigned int row_stride_elems) {
  unsigned long long ga = (unsigned long long)(uintptr_t)gsrc;
  u32x4 g0;
  g0[0] = 1u;                                         // count=1 (valid D#)
  g0[1] = (unsigned int)(uintptr_t)lds_dst;           // lds_addr (bytes)
  g0[2] = (unsigned int)(ga & 0xFFFFFFFFull);         // global_addr[31:0]
  g0[3] = (unsigned int)((ga >> 32) & 0x1FFFFFFull)   // global_addr[56:32]
        | (2u << 30);                                 // type=2 ("image")
  i32x8 g1;
  g1[0] = (int)((2u << 16)        // data_size = 4 bytes
              | (1u << 20)        // pad_enable
              | (4u << 22)        // pad_interval: 32 DWORDs
              | (3u << 25));      // pad_amount: 4 DWORDs
  g1[1] = (int)(tensor_k << 16);                                  // dim0[15:0]
  g1[2] = (int)(((tensor_k >> 16) & 0xFFFFu) |
                ((tensor_rows & 0xFFFFu) << 16));                 // dim0 hi | dim1 lo
  g1[3] = (int)(((tensor_rows >> 16) & 0xFFFFu) |
                ((unsigned)tile_k << 16));                        // dim1 hi | tile_dim0
  g1[4] = (int)((unsigned)tile_rows & 0xFFFFu);                   // tile_dim1 (dim2=0)
  g1[5] = (int)row_stride_elems;                                  // dim0_stride lo32
  g1[6] = 0;                                                      // stride hi / dim1_stride
  g1[7] = 0;
  i32x4 z4 = {0, 0, 0, 0};
#if defined(__clang_major__) && (__clang_major__ >= 23)
  i32x8 z8 = {0, 0, 0, 0, 0, 0, 0, 0};
  __builtin_amdgcn_tensor_load_to_lds(g0, g1, z4, z4, z8, 0);
#else
  __builtin_amdgcn_tensor_load_to_lds(g0, g1, z4, z4, 0);
#endif
}

// =====================================================================
// GEMM: Out[m][n] = sum_k Arow(m)[k] * Bmat[n][k] + bias0[n] (+bias1[n])
//   64x64 tile / block, 8 wave32s, 2 C-tiles per wave, double-buffered.
//   B tile (and A tile when rows are contiguous) staged by the TDM;
//   gathered A rows staged with per-lane async global->LDS copies.
//   f32 -> bf16 with native converts at fragment build; accumulate with
//   v_wmma_f32_16x16x32_bf16.
//   blockIdx.x = M-tile (fast) so blocks sharing a B tile are adjacent in
//   dispatch order -> W_out streams from HBM once (~412 MB total).
// =====================================================================
template <bool GATHER>
__global__ __launch_bounds__(256)
void gemm_bf16_wmma(const float* __restrict__ Abase,
                    const int*   __restrict__ gather, int K,
                    const float* __restrict__ Bmat,   int N,
                    const float* __restrict__ bias0,
                    const float* __restrict__ bias1,
                    float* __restrict__ Out, int ldo)
{
  __shared__ __align__(16) float As[2][64][36];
  __shared__ __align__(16) float Bs[2][64][36];
  __shared__ int rowsrc[64];

  const int tid  = threadIdx.x;
  const int lane = tid & 31;
  const int wave = tid >> 5;
  const int m0   = blockIdx.x * 64;   // M fast
  const int n0   = blockIdx.y * 64;   // N slow

  if (GATHER) {
    if (tid < 64) rowsrc[tid] = gather[m0 + tid];
    __syncthreads();
  }

  const int nc  = wave & 3;
  const int mr0 = (wave >> 2) * 2;
  v8f acc0 = {};
  v8f acc1 = {};

  const int nch = K >> 5;

  // ---- staging helpers ----
  auto stage = [&](int ch, int buf) {
    const int k0 = ch << 5;
    if (GATHER) {
      // gathered rows: per-lane async copies (2 x 16B per thread)
#pragma unroll
      for (int j = 0; j < 2; ++j) {
        const int cid = tid + (j << 8);
        const int row = cid >> 3;
        const int cho = cid & 7;
        const float* src = Abase + (size_t)rowsrc[row] * K + k0 + cho * 4;
        g2l_16B(&As[buf][row][cho * 4], src);
      }
      if (wave == 0)
        tdm_load_2d(&Bs[buf][0][0], Bmat + (size_t)n0 * K + k0,
                    32, 64, (unsigned)K, (unsigned)(N - n0), (unsigned)K);
    } else {
      if (wave == 0) {
        tdm_load_2d(&As[buf][0][0], Abase + (size_t)m0 * K + k0,
                    32, 64, (unsigned)K, 64u, (unsigned)K);
        tdm_load_2d(&Bs[buf][0][0], Bmat + (size_t)n0 * K + k0,
                    32, 64, (unsigned)K, (unsigned)(N - n0), (unsigned)K);
      }
    }
  };

  stage(0, 0);                                  // prologue

  for (int ch = 0; ch < nch; ++ch) {
    const int buf  = ch & 1;
    const bool more = (ch + 1) < nch;
    if (more) stage(ch + 1, buf ^ 1);           // prefetch next chunk

    if (GATHER) {                               // per-wave async copies
      if (more) wait_async_le2(); else wait_async_0();
    }
    if (wave == 0) {                            // wave0's TDM ops
      if (more) __builtin_amdgcn_s_wait_tensorcnt(GATHER ? 1 : 2);
      else      __builtin_amdgcn_s_wait_tensorcnt(0);
    }
    __syncthreads();

    const float (*Asb)[36] = As[buf];
    const float (*Bsb)[36] = Bs[buf];

    // B fragment: lane -> n = nc*16 + lane%16, k = (lane/16)*16 + e
    v16bf bfr;
    {
      const int nl = nc * 16 + (lane & 15);
      const int kb = (lane >> 4) * 16;
      const float4* p = (const float4*)&Bsb[nl][kb];
#pragma unroll
      for (int q = 0; q < 4; ++q) {
        const float4 f = p[q];
        bfr[q * 4 + 0] = (__bf16)f.x;
        bfr[q * 4 + 1] = (__bf16)f.y;
        bfr[q * 4 + 2] = (__bf16)f.z;
        bfr[q * 4 + 3] = (__bf16)f.w;
      }
    }

    // A fragments + WMMA: K run [kb,kb+16), kb=(lane/16)*16
#pragma unroll
    for (int half = 0; half < 2; ++half) {
      const int ml = (mr0 + half) * 16 + (lane & 15);
      const int kb = (lane >> 4) * 16;
      const float4* p0 = (const float4*)&Asb[ml][kb];
      const float4* p1 = (const float4*)&Asb[ml][kb + 16];
      const float4 fa = p0[0], fb = p0[1], fc = p0[2], fd = p0[3];
      v16bf afr;
      afr[0]  = (__bf16)fa.x; afr[1]  = (__bf16)fa.y;
      afr[2]  = (__bf16)fa.z; afr[3]  = (__bf16)fa.w;
      afr[4]  = (__bf16)fb.x; afr[5]  = (__bf16)fb.y;
      afr[6]  = (__bf16)fb.z; afr[7]  = (__bf16)fb.w;
      afr[8]  = (__bf16)fc.x; afr[9]  = (__bf16)fc.y;
      afr[10] = (__bf16)fc.z; afr[11] = (__bf16)fc.w;
      afr[12] = (__bf16)fd.x; afr[13] = (__bf16)fd.y;
      afr[14] = (__bf16)fd.z; afr[15] = (__bf16)fd.w;

      if (half == 0)
        acc0 = __builtin_amdgcn_wmma_f32_16x16x32_bf16(false, afr, false, bfr,
                                                       (short)0, acc0, false, false);
      else
        acc1 = __builtin_amdgcn_wmma_f32_16x16x32_bf16(false, afr, false, bfr,
                                                       (short)0, acc1, false, false);
    }
    __syncthreads();
  }

  // store: C/D layout lane L, vgpr r -> m = r + 8*(L/16), n = L%16
#pragma unroll
  for (int half = 0; half < 2; ++half) {
    F32x8 c; c.v = half ? acc1 : acc0;
    const int nl = n0 + nc * 16 + (lane & 15);
    if (nl < N) {
      const float bb = bias0[nl] + (bias1 ? bias1[nl] : 0.0f);
      const int mbase = m0 + (mr0 + half) * 16 + ((lane >> 4) * 8);
#pragma unroll
      for (int r = 0; r < 8; ++r)
        Out[(size_t)(mbase + r) * ldo + nl] = c.f[r] + bb;
    }
  }
}

// =====================================================================
// Persistent LSTM recurrence: 128 WGs x 256 threads, grid-sync per step.
//   WG w owns h-slice k in [w*8, w*8+8): its 32 W_hh rows (i,f,g,o) are
//   pinned in LDS (1040-float pitch -> conflict-free banking), so each
//   step only moves the 4 KB h vector through L2.
// =====================================================================
#define NWG 128
#define KPW 8

__device__ __forceinline__ float sigf(float x) { return 1.0f / (1.0f + expf(-x)); }

__device__ __forceinline__ void gridsync(unsigned int* c, unsigned int nw) {
  __threadfence();
  __syncthreads();
  if (threadIdx.x == 0) {
    __hip_atomic_fetch_add(c, 1u, __ATOMIC_ACQ_REL, __HIP_MEMORY_SCOPE_AGENT);
    while (__hip_atomic_load(c, __ATOMIC_ACQUIRE, __HIP_MEMORY_SCOPE_AGENT) < nw)
      __builtin_amdgcn_s_sleep(1);
  }
  __syncthreads();
}

__global__ __launch_bounds__(256)
void lstm_recurrence(const float* __restrict__ pre,   // [T][4H]
                     const float* __restrict__ Whh,   // [4H][H]
                     const float* __restrict__ hn,    // [H]
                     const float* __restrict__ cn,    // [H]
                     float* __restrict__ Hseq,        // [(T+1)][H]
                     float* __restrict__ tail,        // d_out + T*V : hT|cT
                     unsigned int* __restrict__ cnt)  // zeroed counters
{
  __shared__ __align__(16) float Wsl[32][1040];  // 133 KB of 320 KB WGP LDS
  __shared__ float h_lds[Hx];
  __shared__ float red[32][8];
  __shared__ float gl[32];
  __shared__ float c_l[KPW];

  const int tid   = threadIdx.x;
  const int kbase = blockIdx.x * KPW;
  const int r     = tid >> 3;
  const int sub   = tid & 7;

  for (int idx = tid; idx < 32 * Hx; idx += 256) {
    const int row  = idx >> 10;
    const int j    = idx & (Hx - 1);
    const int wrow = (row >> 3) * Hx + kbase + (row & 7);
    Wsl[row][j] = Whh[(size_t)wrow * Hx + j];
  }
  if (tid < KPW) {
    c_l[tid] = cn[kbase + tid];
    Hseq[kbase + tid] = hn[kbase + tid];     // H_seq row 0 = h0
  }
  gridsync(&cnt[0], NWG);

  for (int t = 0; t < Tx; ++t) {
    for (int i = tid; i < Hx; i += 256) h_lds[i] = Hseq[(size_t)t * Hx + i];
    __syncthreads();

    float partial = 0.0f;
    const float* wr = &Wsl[r][0];
#pragma unroll 8
    for (int i = 0; i < 128; ++i) {
      const int j = i * 8 + sub;
      partial = fmaf(wr[j], h_lds[j], partial);
    }
    red[r][sub] = partial;
    __syncthreads();

    if (tid < 32) {
      float s = 0.0f;
#pragma unroll
      for (int q = 0; q < 8; ++q) s += red[tid][q];
      const int gate = tid >> 3;
      const int kk   = kbase + (tid & 7);
      gl[tid] = s + pre[(size_t)t * G4H + gate * Hx + kk];
    }
    __syncthreads();

    if (tid < KPW) {
      const float iv = sigf(gl[tid]);
      const float fv = sigf(gl[8 + tid]);
      const float gv = tanhf(gl[16 + tid]);
      const float ov = sigf(gl[24 + tid]);
      const float c  = fv * c_l[tid] + iv * gv;
      c_l[tid] = c;
      const float h = ov * tanhf(c);
      Hseq[(size_t)(t + 1) * Hx + kbase + tid] = h;
      if (t == Tx - 1) {
        tail[kbase + tid]      = h;   // hT
        tail[Hx + kbase + tid] = c;   // cT
      }
    }
    gridsync(&cnt[t + 1], NWG);
  }
}

// =====================================================================
// Host launcher
// =====================================================================
extern "C" void kernel_launch(void* const* d_in, const int* in_sizes, int n_in,
                              void* d_out, int out_size, void* d_ws, size_t ws_size,
                              hipStream_t stream)
{
  (void)in_sizes; (void)n_in; (void)out_size; (void)ws_size;
  const int*   x     = (const int*)  d_in[0];
  const float* hn    = (const float*)d_in[1];
  const float* cn    = (const float*)d_in[2];
  const float* emb   = (const float*)d_in[3];
  const float* W_ih  = (const float*)d_in[4];
  const float* W_hh  = (const float*)d_in[5];
  const float* b_ih  = (const float*)d_in[6];
  const float* b_hh  = (const float*)d_in[7];
  const float* W_out = (const float*)d_in[8];
  const float* b_out = (const float*)d_in[9];
  float* out = (float*)d_out;

  // Workspace: pre [T][4H] | Hseq [(T+1)][H] | cnt [T+2]   (~21 MB)
  char* ws = (char*)d_ws;
  float* pre  = (float*)ws;
  float* Hseq = (float*)(ws + (size_t)Tx * G4H * sizeof(float));
  unsigned int* cnt = (unsigned int*)(ws + (size_t)Tx * G4H * sizeof(float)
                                         + (size_t)(Tx + 1) * Hx * sizeof(float));

  hipMemsetAsync(cnt, 0, (Tx + 2) * sizeof(unsigned int), stream);

  const dim3 blk(256);
  // 1) pre = gather(emb, x) @ W_ih^T + (b_ih + b_hh)    [1024 x 4096]
  gemm_bf16_wmma<true><<<dim3(Tx / 64, G4H / 64), blk, 0, stream>>>(
      emb, x, Ex, W_ih, G4H, b_ih, b_hh, pre, G4H);

  // 2) sequential LSTM scan (persistent, grid-synced)
  lstm_recurrence<<<dim3(NWG), blk, 0, stream>>>(
      pre, W_hh, hn, cn, Hseq, out + (size_t)Tx * Vx, cnt);

  // 3) out = hs @ W_out^T + b_out                       [1024 x 50257]
  gemm_bf16_wmma<false><<<dim3(Tx / 64, (Vx + 63) / 64), blk, 0, stream>>>(
      Hseq + Hx, nullptr, Hx, W_out, Vx, b_out, nullptr, out, Vx);
}
